// Model_89661737271516
// MI455X (gfx1250) — matrix operand order repaired and checked by reading, not compiled, not run on hardware
//
#include <hip/hip_runtime.h>
#include <hip/hip_bf16.h>

// ---------------- problem constants (match reference) ----------------
#define Hh   512
#define H3   1536      // 3*H
#define H2   1024      // 2*H
#define Ss   48
#define Tt   48
#define Bb   32
#define Vout 32000
#define SOS_TOK 1

// ---------------- WMMA vector types (gfx1250, wave32) ----------------
typedef __bf16 bf16_t;
typedef __attribute__((ext_vector_type(16))) __bf16 v16bf;
typedef __attribute__((ext_vector_type(8)))  __bf16 v8bf;
typedef __attribute__((ext_vector_type(8)))  float  v8f;

__device__ __forceinline__ v16bf load_afrag(const bf16_t* row, int k0, int aoff) {
    // A (16x32, MxK) fragment per ISA 7.12.2: lane<16 -> elems 0..7 = K(k0..k0+7),
    // elems 8..15 = K(k0+16..k0+23); lanes 16-31 use +8 K offset (aoff).
    v8bf a0 = *(const v8bf*)(row + k0 + aoff);
    v8bf a1 = *(const v8bf*)(row + k0 + aoff + 16);
    v16bf f;
#pragma unroll
    for (int i = 0; i < 8; ++i) { f[i] = a0[i]; f[i + 8] = a1[i]; }
    return f;
}

__device__ __forceinline__ v16bf load_bfrag(const bf16_t* row, int k0, int boff) {
    // B (32x16, KxN): lane carries one W row (output column); elems 0..7 = K(k0+boff..+7),
    // elems 8..15 = K(k0+boff+16..+23), matching the A fragment K order.
    v8bf b0 = *(const v8bf*)(row + k0 + boff);
    v8bf b1 = *(const v8bf*)(row + k0 + boff + 16);
    v16bf f;
#pragma unroll
    for (int i = 0; i < 8; ++i) { f[i] = b0[i]; f[i + 8] = b1[i]; }
    return f;
}

__device__ __forceinline__ v8f wmma_bf16(v16bf a, v16bf b, v8f c) {
    return __builtin_amdgcn_wmma_f32_16x16x32_bf16(
        /*neg_a=*/false, a, /*neg_b=*/false, b,
        /*c_mod=*/(short)0, c, /*reuse_a=*/false, /*reuse_b=*/false);
}

// ======================================================================
// WMMA GEMM, 2x2 register-blocked: each wave32 owns a 32x32 macro-tile of
//   C[M,N](f32) = A[M,K](bf16) @ W[N,K]^T(bf16) + bias[N]
// Per K-step: 2 A-frag + 2 B-frag loads feed 4 v_wmma_f32_16x16x32_bf16.
// M,N must be multiples of 32; K multiple of 32. nt=1 -> non-temporal C
// stores (streaming logits; keep Wo resident in the 192MB L2 instead).
// ======================================================================
__global__ __launch_bounds__(128) void k_wmma_gemm_bias(
    const bf16_t* __restrict__ A, int lda,
    const bf16_t* __restrict__ W, int ldw,
    const float* __restrict__ bias,
    float* __restrict__ C, int ldc,
    int K, int tilesN, int numTiles, int nt)
{
    int wave = threadIdx.x >> 5;
    int lane = threadIdx.x & 31;
    int tile = blockIdx.x * 4 + wave;
    if (tile >= numTiles) return;           // wave-uniform: EXEC stays all-ones

    int tm   = tile / tilesN;               // 32-row macro tile
    int tn   = tile - tm * tilesN;          // 32-col macro tile
    int half = lane >> 4;                   // 0 = lanes 0-15, 1 = lanes 16-31
    int l16  = lane & 15;

    const bf16_t* arow0 = A + (size_t)(tm * 32 + l16) * (size_t)lda;
    const bf16_t* arow1 = arow0 + (size_t)16 * (size_t)lda;
    const bf16_t* wrow0 = W + (size_t)(tn * 32 + l16) * (size_t)ldw;
    const bf16_t* wrow1 = wrow0 + (size_t)16 * (size_t)ldw;
    int n0 = tn * 32 + l16;
    int n1 = n0 + 16;

    float bv0 = bias ? bias[n0] : 0.0f;
    float bv1 = bias ? bias[n1] : 0.0f;
    v8f acc00, acc01, acc10, acc11;
#pragma unroll
    for (int i = 0; i < 8; ++i) { acc00[i] = bv0; acc01[i] = bv1; acc10[i] = bv0; acc11[i] = bv1; }

    const int aoff = half * 8;
    const int boff = half * 8;

    for (int k0 = 0; k0 < K; k0 += 32) {
        if (k0 + 32 < K) {                  // global_prefetch_b8 on the weight stream
            __builtin_prefetch(wrow0 + k0 + 32 + boff, 0, 1);
            __builtin_prefetch(wrow1 + k0 + 32 + boff, 0, 1);
        }
        v16bf a0 = load_afrag(arow0, k0, aoff);
        v16bf a1 = load_afrag(arow1, k0, aoff);
        v16bf b0 = load_bfrag(wrow0, k0, boff);
        v16bf b1 = load_bfrag(wrow1, k0, boff);
        acc00 = wmma_bf16(a0, b0, acc00);
        acc01 = wmma_bf16(a0, b1, acc01);
        acc10 = wmma_bf16(a1, b0, acc10);
        acc11 = wmma_bf16(a1, b1, acc11);
    }

    int m0 = tm * 32 + half * 8;            // C rows for VGPR j: m0+j (first 16-row tile)
    int m1 = m0 + 16;
    if (nt) {
#pragma unroll
        for (int i = 0; i < 8; ++i) {
            __builtin_nontemporal_store(acc00[i], &C[(size_t)(m0 + i) * (size_t)ldc + n0]);
            __builtin_nontemporal_store(acc01[i], &C[(size_t)(m0 + i) * (size_t)ldc + n1]);
            __builtin_nontemporal_store(acc10[i], &C[(size_t)(m1 + i) * (size_t)ldc + n0]);
            __builtin_nontemporal_store(acc11[i], &C[(size_t)(m1 + i) * (size_t)ldc + n1]);
        }
    } else {
#pragma unroll
        for (int i = 0; i < 8; ++i) {
            C[(size_t)(m0 + i) * (size_t)ldc + n0] = acc00[i];
            C[(size_t)(m0 + i) * (size_t)ldc + n1] = acc01[i];
            C[(size_t)(m1 + i) * (size_t)ldc + n0] = acc10[i];
            C[(size_t)(m1 + i) * (size_t)ldc + n1] = acc11[i];
        }
    }
}

// ---------------- elementwise / small kernels ----------------
__global__ void k_f32_to_bf16(const float* __restrict__ src,
                              __hip_bfloat16* __restrict__ dst, int n) {
    int i = blockIdx.x * blockDim.x + threadIdx.x;
    if (i < n) dst[i] = __float2bfloat16(src[i]);
}

__global__ void k_embed(const int* __restrict__ toks, const float* __restrict__ emb,
                        __hip_bfloat16* __restrict__ out, int nrows) {
    int i = blockIdx.x * blockDim.x + threadIdx.x;
    if (i >= nrows * Hh) return;
    int r = i / Hh, j = i - r * Hh;
    out[i] = __float2bfloat16(emb[(size_t)toks[r] * Hh + j]);
}

__global__ void k_dec_embed(const int* __restrict__ target, int t,
                            const float* __restrict__ emb,
                            __hip_bfloat16* __restrict__ out) {
    int i = blockIdx.x * blockDim.x + threadIdx.x;
    if (i >= Bb * Hh) return;
    int b = i / Hh, j = i - b * Hh;
    int tok = (t == 0) ? SOS_TOK : target[(t - 1) * Bb + b];
    out[i] = __float2bfloat16(emb[(size_t)tok * Hh + j]);
}

__device__ __forceinline__ float sigm(float x) { return 1.0f / (1.0f + __expf(-x)); }

// GRU gate fusion: h' = (1-z)*n + z*h; mode 0: none, 1: store h' to out, 2: add
__global__ void k_gru_gate(const float* __restrict__ gi, const float* __restrict__ gh,
                           float* __restrict__ h, __hip_bfloat16* __restrict__ hbf,
                           float* __restrict__ out_slice, int mode) {
    int i = blockIdx.x * blockDim.x + threadIdx.x;
    if (i >= Bb * Hh) return;
    int b = i / Hh, j = i - b * Hh;
    const float* gib = gi + (size_t)b * H3;
    const float* ghb = gh + (size_t)b * H3;
    float r  = sigm(gib[j] + ghb[j]);
    float z  = sigm(gib[Hh + j] + ghb[Hh + j]);
    float nn = tanhf(gib[2 * Hh + j] + r * ghb[2 * Hh + j]);
    float hp = h[i];
    float hn = (1.0f - z) * nn + z * hp;
    h[i]   = hn;
    hbf[i] = __float2bfloat16(hn);
    if (mode == 1)      out_slice[i]  = hn;
    else if (mode == 2) out_slice[i] += hn;
}

// Luong dot attention + softmax + context; writes concat[b] = [h2 | ctx] in bf16
__global__ void k_attention(const float* __restrict__ h2,
                            const float* __restrict__ enc_out,
                            __hip_bfloat16* __restrict__ concat) {
    __shared__ float red[256];
    __shared__ float sc[Ss];
    int b = blockIdx.x, t = threadIdx.x;
    for (int s = 0; s < Ss; ++s) {
        float p = 0.0f;
        for (int j = t; j < Hh; j += 256)
            p += h2[b * Hh + j] * enc_out[((size_t)s * Bb + b) * Hh + j];
        red[t] = p; __syncthreads();
        for (int w = 128; w > 0; w >>= 1) { if (t < w) red[t] += red[t + w]; __syncthreads(); }
        if (t == 0) sc[s] = red[0];
        __syncthreads();
    }
    if (t == 0) {
        float m = sc[0];
        for (int s = 1; s < Ss; ++s) m = fmaxf(m, sc[s]);
        float sum = 0.0f;
        for (int s = 0; s < Ss; ++s) { sc[s] = __expf(sc[s] - m); sum += sc[s]; }
        float inv = 1.0f / sum;
        for (int s = 0; s < Ss; ++s) sc[s] *= inv;
    }
    __syncthreads();
    for (int j = t; j < Hh; j += 256) {
        float acc = 0.0f;
        for (int s = 0; s < Ss; ++s) acc += sc[s] * enc_out[((size_t)s * Bb + b) * Hh + j];
        concat[(size_t)b * H2 + j]      = __float2bfloat16(h2[b * Hh + j]);
        concat[(size_t)b * H2 + Hh + j] = __float2bfloat16(acc);
    }
}

__global__ void k_tanh_bf16(const float* __restrict__ in,
                            __hip_bfloat16* __restrict__ out, int n) {
    int i = blockIdx.x * blockDim.x + threadIdx.x;
    if (i < n) out[i] = __float2bfloat16(tanhf(in[i]));
}

// ======================================================================
extern "C" void kernel_launch(void* const* d_in, const int* in_sizes, int n_in,
                              void* d_out, int out_size, void* d_ws, size_t ws_size,
                              hipStream_t stream) {
    (void)in_sizes; (void)n_in; (void)out_size; (void)ws_size;

    // inputs, in setup_inputs() order
    const int*   input_batches  = (const int*)  d_in[0];   // [S,B]
    const int*   target_batches = (const int*)  d_in[1];   // [T,B]
    const float* enc_emb   = (const float*)d_in[2];
    const float* enc_Wih_f = (const float*)d_in[3];
    const float* enc_Whh_f = (const float*)d_in[4];
    const float* enc_bih_f = (const float*)d_in[5];
    const float* enc_bhh_f = (const float*)d_in[6];
    const float* enc_Wih_b = (const float*)d_in[7];
    const float* enc_Whh_b = (const float*)d_in[8];
    const float* enc_bih_b = (const float*)d_in[9];
    const float* enc_bhh_b = (const float*)d_in[10];
    const float* dec_emb   = (const float*)d_in[11];
    const float* dec_Wih   = (const float*)d_in[12];
    const float* dec_Whh   = (const float*)d_in[13];
    const float* dec_bih   = (const float*)d_in[14];
    const float* dec_bhh   = (const float*)d_in[15];
    const float* Wc        = (const float*)d_in[16];
    const float* bc        = (const float*)d_in[17];
    const float* Wo        = (const float*)d_in[18];
    const float* bo        = (const float*)d_in[19];

    float* out = (float*)d_out;                            // [T,B,V]

    // -------- workspace carve-up --------
    char* base = (char*)d_ws;
    size_t off = 0;
    auto carve = [&](size_t bytes) -> char* {
        char* p = base + off;
        off = (off + bytes + 255) & ~(size_t)255;
        return p;
    };
    const size_t WGRU = (size_t)H3 * Hh;                   // 1536*512
    __hip_bfloat16* wWihF = (__hip_bfloat16*)carve(WGRU * 2);
    __hip_bfloat16* wWhhF = (__hip_bfloat16*)carve(WGRU * 2);
    __hip_bfloat16* wWihB = (__hip_bfloat16*)carve(WGRU * 2);
    __hip_bfloat16* wWhhB = (__hip_bfloat16*)carve(WGRU * 2);
    __hip_bfloat16* wDWih = (__hip_bfloat16*)carve(WGRU * 2);
    __hip_bfloat16* wDWhh = (__hip_bfloat16*)carve(WGRU * 2);
    __hip_bfloat16* wWc   = (__hip_bfloat16*)carve((size_t)Hh * H2 * 2);
    __hip_bfloat16* wWo   = (__hip_bfloat16*)carve((size_t)Vout * Hh * 2);
    __hip_bfloat16* x_bf  = (__hip_bfloat16*)carve((size_t)Ss * Bb * Hh * 2);
    float* gi_f   = (float*)carve((size_t)Ss * Bb * H3 * 4);
    float* gi_b   = (float*)carve((size_t)Ss * Bb * H3 * 4);
    float* enc_o  = (float*)carve((size_t)Ss * Bb * Hh * 4);
    float* h      = (float*)carve((size_t)Bb * Hh * 4);    // fwd enc, then decoder hidden
    __hip_bfloat16* h_bf  = (__hip_bfloat16*)carve((size_t)Bb * Hh * 2);
    float* hb     = (float*)carve((size_t)Bb * Hh * 4);    // bwd enc hidden
    __hip_bfloat16* hb_bf = (__hip_bfloat16*)carve((size_t)Bb * Hh * 2);
    float* gh     = (float*)carve((size_t)Bb * H3 * 4);
    float* gi_d   = (float*)carve((size_t)Bb * H3 * 4);
    __hip_bfloat16* e_bf  = (__hip_bfloat16*)carve((size_t)Bb * Hh * 2);
    __hip_bfloat16* cat_bf= (__hip_bfloat16*)carve((size_t)Bb * H2 * 2);
    float* cpre   = (float*)carve((size_t)Bb * Hh * 4);
    __hip_bfloat16* c_bf  = (__hip_bfloat16*)carve((size_t)Bb * Hh * 2);

    auto cvt = [&](const float* s, __hip_bfloat16* d, size_t n) {
        k_f32_to_bf16<<<(int)((n + 255) / 256), 256, 0, stream>>>(s, d, (int)n);
    };
    auto gemm = [&](const __hip_bfloat16* A, int lda, const __hip_bfloat16* Wm, int ldw,
                    const float* bias, float* C, int ldc, int M, int N, int K, int nt) {
        int tilesN = N / 32, tiles = (M / 32) * tilesN;    // 32x32 macro-tiles
        int blocks = (tiles + 3) / 4;
        k_wmma_gemm_bias<<<blocks, 128, 0, stream>>>(
            (const bf16_t*)A, lda, (const bf16_t*)Wm, ldw, bias, C, ldc, K, tilesN, tiles, nt);
    };

    // -------- 1) weights -> bf16 --------
    cvt(enc_Wih_f, wWihF, WGRU);  cvt(enc_Whh_f, wWhhF, WGRU);
    cvt(enc_Wih_b, wWihB, WGRU);  cvt(enc_Whh_b, wWhhB, WGRU);
    cvt(dec_Wih,   wDWih, WGRU);  cvt(dec_Whh,   wDWhh, WGRU);
    cvt(Wc, wWc, (size_t)Hh * H2);
    cvt(Wo, wWo, (size_t)Vout * Hh);

    // -------- 2) encoder --------
    k_embed<<<(Ss * Bb * Hh + 255) / 256, 256, 0, stream>>>(input_batches, enc_emb, x_bf, Ss * Bb);
    // batched input-side GEMMs for all timesteps (no sequential dependence)
    gemm(x_bf, Hh, wWihF, Hh, enc_bih_f, gi_f, H3, Ss * Bb, H3, Hh, 0);
    gemm(x_bf, Hh, wWihB, Hh, enc_bih_b, gi_b, H3, Ss * Bb, H3, Hh, 0);

    hipMemsetAsync(h,     0, (size_t)Bb * Hh * 4, stream);
    hipMemsetAsync(h_bf,  0, (size_t)Bb * Hh * 2, stream);
    hipMemsetAsync(hb,    0, (size_t)Bb * Hh * 4, stream);
    hipMemsetAsync(hb_bf, 0, (size_t)Bb * Hh * 2, stream);

    for (int s = 0; s < Ss; ++s) {            // forward direction
        gemm(h_bf, Hh, wWhhF, Hh, enc_bhh_f, gh, H3, Bb, H3, Hh, 0);
        k_gru_gate<<<(Bb * Hh + 255) / 256, 256, 0, stream>>>(
            gi_f + (size_t)s * Bb * H3, gh, h, h_bf, enc_o + (size_t)s * Bb * Hh, 1);
    }
    for (int s = Ss - 1; s >= 0; --s) {       // backward direction (sums into enc_o)
        gemm(hb_bf, Hh, wWhhB, Hh, enc_bhh_b, gh, H3, Bb, H3, Hh, 0);
        k_gru_gate<<<(Bb * Hh + 255) / 256, 256, 0, stream>>>(
            gi_b + (size_t)s * Bb * H3, gh, hb, hb_bf, enc_o + (size_t)s * Bb * Hh, 2);
    }
    // decoder hidden = final forward state: already in h / h_bf

    // -------- 3) decoder with teacher forcing --------
    for (int t = 0; t < Tt; ++t) {
        k_dec_embed<<<(Bb * Hh + 255) / 256, 256, 0, stream>>>(target_batches, t, dec_emb, e_bf);
        gemm(e_bf, Hh, wDWih, Hh, dec_bih, gi_d, H3, Bb, H3, Hh, 0);
        gemm(h_bf, Hh, wDWhh, Hh, dec_bhh, gh,   H3, Bb, H3, Hh, 0);
        k_gru_gate<<<(Bb * Hh + 255) / 256, 256, 0, stream>>>(gi_d, gh, h, h_bf, nullptr, 0);
        k_attention<<<Bb, 256, 0, stream>>>(h, enc_o, cat_bf);
        gemm(cat_bf, H2, wWc, H2, bc, cpre, Hh, Bb, Hh, H2, 0);
        k_tanh_bf16<<<(Bb * Hh + 255) / 256, 256, 0, stream>>>(cpre, c_bf, Bb * Hh);
        // dominant GEMM: logits = c @ Wo^T + bo -> d_out[t]  (streamed, non-temporal)
        gemm(c_bf, Hh, wWo, Hh, bo, out + (size_t)t * Bb * Vout, Vout, Bb, Vout, Hh, 1);
    }
}
